// GraphTransformerEncoder_81982335746575
// MI455X (gfx1250) — matrix operand order repaired and checked by reading, not compiled, not run on hardware
//
#include <hip/hip_runtime.h>
#include <hip/hip_bf16.h>

#define NN  50000
#define EE  800000
#define DD  128
#define HH  4
#define DHH 32

#define MTILES (NN / 16)          // 3125
#define AFRAGS (MTILES * 4)       // 12500 fragments per input matrix
#define FRAGSZ 512                // 32 lanes x 16 bf16 elements
#define WMATSZ (DD * DD)          // 16384 = 32 fragments x 512

typedef __bf16 bf16_t;
typedef __attribute__((ext_vector_type(16))) __bf16 v16bf;
typedef __attribute__((ext_vector_type(8)))  float  v8f;

// ---------- helpers: orderable-uint encoding for float atomic max ----------
__device__ __forceinline__ unsigned f2key(float f) {
    unsigned u = __float_as_uint(f);
    return (u & 0x80000000u) ? ~u : (u | 0x80000000u);
}
__device__ __forceinline__ float key2f(unsigned k) {
    return (k & 0x80000000u) ? __uint_as_float(k ^ 0x80000000u)
                             : __uint_as_float(~k);
}

// ---------- pack activation matrix into A-fragment layout (f32 -> bf16) ----------
// fragment id = mt*4 + kk; per lane 16 elements stored contiguously (32B).
// A 16x32 bf16 ISA layout: lane<16 K={0..7,16..23}, lane>=16 K={8..15,24..31}.
__global__ void pack_a(const float* __restrict__ x, bf16_t* __restrict__ apack, int relu) {
    const int frag = (blockIdx.x * blockDim.x + threadIdx.x) >> 5;
    if (frag >= AFRAGS) return;
    const int lane = threadIdx.x & 31;
    const int mt = frag >> 2, kk = frag & 3;
    const int hi = lane >> 4, l15 = lane & 15;
    const int row = mt * 16 + l15;
    const int kb  = kk * 32;
    bf16_t* dst = apack + ((size_t)frag * 32 + lane) * 16;
#pragma unroll
    for (int i = 0; i < 16; ++i) {
        const int r = i >> 1, p = i & 1;
        const int ka = ((r & 4) ? 16 : 0) + (hi ? 8 : 0) + ((r & 3) << 1) + p;
        float f = x[(size_t)row * DD + kb + ka];
        if (relu) f = f > 0.0f ? f : 0.0f;
        dst[i] = (bf16_t)f;
    }
}

// ---------- pack one weight matrix into B-fragment layout (f32 -> bf16) ----------
// fragment id = kk*8 + nt; B 32x16: lane<16 K={0..7,16..23}, lane>=16 K={8..15,24..31}.
__global__ void pack_b(const float* __restrict__ w, bf16_t* __restrict__ out) {
    const int frag = (blockIdx.x * blockDim.x + threadIdx.x) >> 5;
    if (frag >= 32) return;
    const int lane = threadIdx.x & 31;
    const int kk = frag >> 3, nt = frag & 7;
    const int hi = lane >> 4, l15 = lane & 15;
    const int col = nt * 16 + l15;
    const int kb  = kk * 32 + (hi ? 8 : 0);
    bf16_t* dst = out + ((size_t)frag * 32 + lane) * 16;
#pragma unroll
    for (int i = 0; i < 16; ++i)
        dst[i] = (bf16_t)w[(size_t)(kb + (i & 7) + ((i & 8) ? 16 : 0)) * DD + col];
}

// ---------- fused 4-way projection GEMM:  {q,k,v,skip} = x @ w + b ----------
// One wave per 16x16 output tile; packed fragments -> pure b128 loads + wmma.
// 25000 waves exactly (3125 blocks x 8 waves), EXEC all-ones throughout.
__global__ void proj4_wmma(const bf16_t* __restrict__ apack,
                           const bf16_t* __restrict__ bpack,  // 4 packed matrices (q,k,v,s)
                           const float* __restrict__ bq, const float* __restrict__ bk,
                           const float* __restrict__ bv, const float* __restrict__ bs,
                           float* __restrict__ oq, float* __restrict__ ok,
                           float* __restrict__ ov, float* __restrict__ os) {
    const int wave = (blockIdx.x * blockDim.x + threadIdx.x) >> 5;
    const int lane = threadIdx.x & 31;
    const int mt   = wave >> 3;       // row tile 0..3124
    const int nt   = wave & 7;        // col tile 0..7
    const int hi   = lane >> 4;
    const int l15  = lane & 15;
    const int col  = nt * 16 + l15;

    v8f acc0 = {}, acc1 = {}, acc2 = {}, acc3 = {};

#pragma unroll
    for (int kk = 0; kk < 4; ++kk) {
        const v16bf a = *(const v16bf*)(apack + ((size_t)(mt * 4 + kk) * 32 + lane) * 16);
        const bf16_t* bb = bpack + ((size_t)(kk * 8 + nt) * 32 + lane) * 16;
        const v16bf b0 = *(const v16bf*)(bb + 0 * (size_t)WMATSZ);
        const v16bf b1 = *(const v16bf*)(bb + 1 * (size_t)WMATSZ);
        const v16bf b2 = *(const v16bf*)(bb + 2 * (size_t)WMATSZ);
        const v16bf b3 = *(const v16bf*)(bb + 3 * (size_t)WMATSZ);
        acc0 = __builtin_amdgcn_wmma_f32_16x16x32_bf16(false, a, false, b0, (short)0, acc0, false, false);
        acc1 = __builtin_amdgcn_wmma_f32_16x16x32_bf16(false, a, false, b1, (short)0, acc1, false, false);
        acc2 = __builtin_amdgcn_wmma_f32_16x16x32_bf16(false, a, false, b2, (short)0, acc2, false, false);
        acc3 = __builtin_amdgcn_wmma_f32_16x16x32_bf16(false, a, false, b3, (short)0, acc3, false, false);
    }

    const float biq = bq[col], bik = bk[col], biv = bv[col], bis = bs[col];
#pragma unroll
    for (int r = 0; r < 8; ++r) {
        const int orow = mt * 16 + r + hi * 8;   // C/D layout: VGPR r -> row r (+8 hi half)
        const size_t oi = (size_t)orow * DD + col;
        oq[oi] = acc0[r] + biq;
        ok[oi] = acc1[r] + bik;
        ov[oi] = acc2[r] + biv;
        os[oi] = acc3[r] + bis;
    }
}

// ---------- per-edge attention logits + running segment max ----------
// one wave per edge; lane = dim-in-head, head j = lane-group (D = 4*32)
__global__ void edge_logits(const long long* __restrict__ src, const long long* __restrict__ dst,
                            const float* __restrict__ q, const float* __restrict__ k,
                            float* __restrict__ logits, unsigned* __restrict__ maxbuf) {
    const int e = (blockIdx.x * blockDim.x + threadIdx.x) >> 5;
    if (e >= EE) return;
    const int lane = threadIdx.x & 31;
    const long long s = src[e];
    const long long d = dst[e];
    const float* qrow = q + (size_t)d * DD;
    const float* krow = k + (size_t)s * DD;

    float p0 = qrow[0 * DHH + lane] * krow[0 * DHH + lane];
    float p1 = qrow[1 * DHH + lane] * krow[1 * DHH + lane];
    float p2 = qrow[2 * DHH + lane] * krow[2 * DHH + lane];
    float p3 = qrow[3 * DHH + lane] * krow[3 * DHH + lane];
#pragma unroll
    for (int off = 16; off > 0; off >>= 1) {
        p0 += __shfl_xor(p0, off, 32);
        p1 += __shfl_xor(p1, off, 32);
        p2 += __shfl_xor(p2, off, 32);
        p3 += __shfl_xor(p3, off, 32);
    }
    if (lane < HH) {
        float val = (lane == 0) ? p0 : (lane == 1) ? p1 : (lane == 2) ? p2 : p3;
        val *= 0.17677669529663687f;  // 1/sqrt(Dh)
        logits[(size_t)e * HH + lane] = val;
        atomicMax(&maxbuf[(size_t)d * HH + lane], f2key(val));
    }
}

// ---------- exp(logit - segmax) + segment sum ----------
__global__ void edge_expsum(const long long* __restrict__ dst,
                            float* __restrict__ logits,
                            const unsigned* __restrict__ maxbuf,
                            float* __restrict__ sumbuf) {
    const int idx = blockIdx.x * blockDim.x + threadIdx.x;
    if (idx >= EE * HH) return;
    const int e = idx >> 2, h = idx & 3;
    const long long d = dst[e];
    const float m  = key2f(maxbuf[(size_t)d * HH + h]);
    const float ev = __expf(logits[idx] - m);
    logits[idx] = ev;  // buffer now holds exp values
    atomicAdd(&sumbuf[(size_t)d * HH + h], ev);
}

// ---------- normalized message aggregation (scatter-add into skip buffer) ----------
__global__ void edge_agg(const long long* __restrict__ src, const long long* __restrict__ dst,
                         const float* __restrict__ v, const float* __restrict__ expv,
                         const float* __restrict__ sumbuf, float* __restrict__ out) {
    const int e = (blockIdx.x * blockDim.x + threadIdx.x) >> 5;
    if (e >= EE) return;
    const int lane = threadIdx.x & 31;
    const long long s = src[e];
    const long long d = dst[e];
#pragma unroll
    for (int j = 0; j < HH; ++j) {
        const float a   = expv[(size_t)e * HH + j] / (sumbuf[(size_t)d * HH + j] + 1e-16f);
        const float val = v[(size_t)s * DD + j * DHH + lane] * a;
        atomicAdd(&out[(size_t)d * DD + j * DHH + lane], val);
    }
}

extern "C" void kernel_launch(void* const* d_in, const int* in_sizes, int n_in,
                              void* d_out, int out_size, void* d_ws, size_t ws_size,
                              hipStream_t stream) {
    (void)in_sizes; (void)n_in; (void)out_size; (void)ws_size;

    const float*     x    = (const float*)d_in[0];
    const long long* eidx = (const long long*)d_in[1];   // int64 [2,E]
    const long long* srcp = eidx;
    const long long* dstp = eidx + EE;
    const float* w[8]; const float* b[8];
    for (int i = 0; i < 8; ++i) {                         // order: l0 q,k,v,s ; l1 q,k,v,s
        w[i] = (const float*)d_in[2 + 2 * i];
        b[i] = (const float*)d_in[3 + 2 * i];
    }

    char* ws = (char*)d_ws;
    size_t off = 0;
    auto take = [&](size_t bytes) -> char* {
        char* p = ws + off;
        off = (off + bytes + 255) & ~(size_t)255;
        return p;
    };
    bf16_t*   apack = (bf16_t*)take((size_t)AFRAGS * FRAGSZ * 2);   // 12.8 MB, reused by both layers
    bf16_t*   bpack = (bf16_t*)take((size_t)8 * WMATSZ * 2);        // 256 KB
    float*    qb    = (float*)take((size_t)NN * DD * 4);
    float*    kb    = (float*)take((size_t)NN * DD * 4);
    float*    vb    = (float*)take((size_t)NN * DD * 4);
    float*    sk    = (float*)take((size_t)NN * DD * 4);
    float*    lg    = (float*)take((size_t)EE * HH * 4);
    unsigned* mx    = (unsigned*)take((size_t)NN * HH * 4);
    float*    sm    = (float*)take((size_t)NN * HH * 4);

    // one-time weight packing into B-fragment layout (bf16)
    for (int i = 0; i < 8; ++i)
        pack_b<<<4, 256, 0, stream>>>(w[i], bpack + (size_t)i * WMATSZ);

    const int packABlocks = (AFRAGS * 32 + 255) / 256;   // 1563
    const int projBlocks  = (MTILES * 8) / 8;            // 3125 blocks x 8 waves
    const int edgeBlocks  = EE / 8;                      // 100000 blocks (wave per edge)
    float* outp = (float*)d_out;

    // ----- layer 0 -----
    pack_a<<<packABlocks, 256, 0, stream>>>(x, apack, 0);
    proj4_wmma<<<projBlocks, 256, 0, stream>>>(
        apack, bpack + 0 * (size_t)WMATSZ,
        b[0], b[1], b[2], b[3], qb, kb, vb, sk);
    hipMemsetAsync(mx, 0, (size_t)NN * HH * 4, stream);
    hipMemsetAsync(sm, 0, (size_t)NN * HH * 4, stream);
    edge_logits<<<edgeBlocks, 256, 0, stream>>>(srcp, dstp, qb, kb, lg, mx);
    edge_expsum<<<(EE * HH) / 256, 256, 0, stream>>>(dstp, lg, mx, sm);
    edge_agg<<<edgeBlocks, 256, 0, stream>>>(srcp, dstp, vb, lg, sm, sk);

    // ----- layer 1 (ReLU fused into A-packing; skip written into d_out) -----
    pack_a<<<packABlocks, 256, 0, stream>>>(sk, apack, 1);
    proj4_wmma<<<projBlocks, 256, 0, stream>>>(
        apack, bpack + 4 * (size_t)WMATSZ,
        b[4], b[5], b[6], b[7], qb, kb, vb, outp);
    hipMemsetAsync(mx, 0, (size_t)NN * HH * 4, stream);
    hipMemsetAsync(sm, 0, (size_t)NN * HH * 4, stream);
    edge_logits<<<edgeBlocks, 256, 0, stream>>>(srcp, dstp, qb, kb, lg, mx);
    edge_expsum<<<(EE * HH) / 256, 256, 0, stream>>>(dstp, lg, mx, sm);
    edge_agg<<<edgeBlocks, 256, 0, stream>>>(srcp, dstp, vb, lg, sm, outp);
}
